// LSTM_64012192579728
// MI455X (gfx1250) — matrix operand order repaired and checked by reading, not compiled, not run on hardware
//
#include <hip/hip_runtime.h>
#include <hip/hip_bf16.h>
#include <math.h>

typedef __attribute__((ext_vector_type(16))) __bf16 v16bf;
typedef __attribute__((ext_vector_type(8)))  float  v8f;

#define T_STEPS 1024
#define BATCH   64
#define EDIM    512
#define HDIM    512
#define G4H     2048
#define NWG     16      // workgroups for recurrent kernel
#define RTHREADS 512    // 16 waves
// LDS pitches (bf16 elements): 520*2 = 1040 bytes = 65*16 -> 16B aligned rows, conflict-free
#define WPITCH  520
#define HPITCH  520

__device__ inline v8f wmma_bf16(v16bf a, v16bf b, v8f c) {
  return __builtin_amdgcn_wmma_f32_16x16x32_bf16(false, a, false, b, (short)0, c, false, false);
}

// CDNA5 async global->LDS copy (ASYNCcnt-tracked, bypasses VGPRs).
// lds_off: byte offset within LDS (low 32 bits of a generic pointer to shared mem).
__device__ inline void async_g2l_b128(unsigned lds_off, const void* gaddr) {
  asm volatile("global_load_async_to_lds_b128 %0, %1, off"
               :: "v"(lds_off), "v"(gaddr)
               : "memory");
}
__device__ inline void wait_asynccnt0() {
  asm volatile("s_wait_asynccnt 0x0" ::: "memory");
}

// A fragment: 16x32 bf16, row-major LDS with pitch. Lane r (0..15), hb = lane>>4.
// VGPR v0..3: K = hb*8 + {0,2,4,6}(+1); v4..7: K = 16 + hb*8 + {0,2,4,6}(+1)
__device__ inline v16bf load_frag_a(const __bf16* base, int pitch, int row0, int k0,
                                    int r, int hb) {
  union { unsigned u[8]; v16bf v; } f;
  const __bf16* p = base + (size_t)(row0 + r) * pitch + k0 + hb * 8;
#pragma unroll
  for (int v = 0; v < 4; ++v) f.u[v]     = *(const unsigned*)(p + 2 * v);
#pragma unroll
  for (int v = 0; v < 4; ++v) f.u[4 + v] = *(const unsigned*)(p + 16 + 2 * v);
  return f.v;
}

// B fragment: 32x16 bf16 from column-major LDS (ws[n*pitch + k]).
// Lane r -> col: v0..3: K = hb*8 + 2v (+1); v4..7: K = 16 + hb*8 + 2(v-4) (+1).
__device__ inline v16bf load_frag_b(const __bf16* base, int pitch, int col0, int k0,
                                    int r, int hb) {
  union { unsigned u[8]; v16bf v; } f;
  const __bf16* p = base + (size_t)(col0 + r) * pitch + k0 + hb * 8;
#pragma unroll
  for (int v = 0; v < 8; ++v) f.u[v] = *(const unsigned*)(p + 2 * v + (v >= 4 ? 8 : 0));
  return f.v;
}

// ---------------- prep: concat/convert weights + bias, zero h & barrier ----------------
__global__ void prep_kernel(const float* __restrict__ wgx, const float* __restrict__ wgh,
                            const float* __restrict__ bg,
                            const float* __restrict__ wix, const float* __restrict__ wih,
                            const float* __restrict__ bi,
                            const float* __restrict__ wfx, const float* __restrict__ wfh,
                            const float* __restrict__ bf_,
                            const float* __restrict__ wox, const float* __restrict__ woh,
                            const float* __restrict__ bo,
                            __bf16* __restrict__ Wx, __bf16* __restrict__ Wh,
                            float* __restrict__ bias_cat, __bf16* __restrict__ h_glob,
                            unsigned* __restrict__ bar) {
  const size_t NW = (size_t)EDIM * G4H;
  size_t idx = (size_t)blockIdx.x * 256 + threadIdx.x;
  const float* xs[4] = {wgx, wix, wfx, wox};
  const float* hs4[4] = {wgh, wih, wfh, woh};
  const float* bs[4] = {bg, bi, bf_, bo};
  if (idx < NW) {
    int e = (int)(idx >> 11);
    int c2 = (int)(idx & 2047);
    int gate = c2 >> 9, c = c2 & 511;
    Wx[idx] = (__bf16)xs[gate][(size_t)e * 512 + c];
    Wh[idx] = (__bf16)hs4[gate][(size_t)e * 512 + c];
  } else if (idx < NW + G4H) {
    int c2 = (int)(idx - NW);
    bias_cat[c2] = bs[c2 >> 9][c2 & 511];
  } else if (idx < NW + G4H + (size_t)BATCH * HDIM) {
    h_glob[idx - NW - G4H] = (__bf16)0.f;
  } else if (idx == NW + G4H + (size_t)BATCH * HDIM) {
    *bar = 0u;
  }
}

// ---------------- x projection: xp[65536,2048] = embeds[65536,512] @ Wx + bias ----------------
__global__ __launch_bounds__(256) void xproj_kernel(const float* __restrict__ embeds,
                                                    const __bf16* __restrict__ Wx,
                                                    const float* __restrict__ bias,
                                                    float* __restrict__ xp) {
  __shared__ __bf16 A_s[128 * 40];  // 128 rows x 32 K, pitch 40
  __shared__ __bf16 B_s[64 * 40];   // 64 cols  x 32 K, col-major, pitch 40
  const int tid = threadIdx.x;
  const int lane = tid & 31, r = lane & 15, hb = lane >> 4;
  const int wave = tid >> 5, wm = wave & 3, wn = wave >> 2;  // 4x2 waves -> 128x64 tile
  const int m0 = blockIdx.y * 128, n0 = blockIdx.x * 64;
  v8f acc[2][2] = {};
  for (int kk = 0; kk < 16; ++kk) {
    for (int i = tid; i < 128 * 32; i += 256) {           // stage A (f32 -> bf16)
      int rr = i >> 5, c = i & 31;
      A_s[rr * 40 + c] = (__bf16)embeds[(size_t)(m0 + rr) * EDIM + kk * 32 + c];
    }
    for (int i = tid; i < 64 * 32; i += 256) {            // stage B col-major
      int n = i & 63, k = i >> 6;
      B_s[n * 40 + k] = Wx[(size_t)(kk * 32 + k) * G4H + n0 + n];
    }
    __syncthreads();
    v16bf a0 = load_frag_a(A_s, 40, wm * 32, 0, r, hb);
    v16bf a1 = load_frag_a(A_s, 40, wm * 32 + 16, 0, r, hb);
    v16bf b0 = load_frag_b(B_s, 40, wn * 32, 0, r, hb);
    v16bf b1 = load_frag_b(B_s, 40, wn * 32 + 16, 0, r, hb);
    acc[0][0] = wmma_bf16(a0, b0, acc[0][0]);
    acc[0][1] = wmma_bf16(a0, b1, acc[0][1]);
    acc[1][0] = wmma_bf16(a1, b0, acc[1][0]);
    acc[1][1] = wmma_bf16(a1, b1, acc[1][1]);
    __syncthreads();
  }
#pragma unroll
  for (int i = 0; i < 2; ++i)
#pragma unroll
    for (int j = 0; j < 2; ++j) {
      int grow_base = m0 + wm * 32 + i * 16 + hb * 8;
      int gcol = n0 + wn * 32 + j * 16 + r;
      float bz = bias[gcol];
#pragma unroll
      for (int g = 0; g < 8; ++g) {
        size_t row = (size_t)(grow_base + g);
        xp[row * G4H + gcol] = acc[i][j][g] + bz;
      }
    }
}

// ---------------- persistent recurrent kernel: 16 WGs, Wh slice resident in LDS ----------------
__global__ __launch_bounds__(RTHREADS) void lstm_rec(const __bf16* __restrict__ Wh,
                                                     const float* __restrict__ xp,
                                                     __bf16* __restrict__ h_glob,
                                                     float* __restrict__ out,
                                                     unsigned* __restrict__ bar) {
  extern __shared__ char smem[];
  __bf16* w_s = (__bf16*)smem;                             // [128][WPITCH] col-major  ~130KB
  __bf16* h_s = (__bf16*)(smem + 128 * WPITCH * 2);        // [64][HPITCH]             ~65KB
  float*  zs  = (float*)(smem + 128 * WPITCH * 2);         // overlay: [4][64][32] f32  32KB
  const int tid = threadIdx.x;
  const int wg = blockIdx.x;                               // owns h columns [wg*32, wg*32+32)
  const int lane = tid & 31, r = lane & 15, hb = lane >> 4;
  const int wave = tid >> 5, mt = wave & 3, ng = wave >> 2; // mt: batch tile, ng: gate

  // load Wh slice once: local col n (0..127) -> global z-col gate*512 + wg*32 + j
  for (int i = tid; i < 128 * 512; i += RTHREADS) {
    int n = i & 127, k = i >> 7;
    int zc = (n >> 5) * HDIM + wg * 32 + (n & 31);
    w_s[n * WPITCH + k] = Wh[(size_t)k * G4H + zc];
  }
  float creg[4] = {0.f, 0.f, 0.f, 0.f};

  // precompute async-staging addresses: thread handles 8 16B chunks of h [64][512]bf16
  // chunk i (0..4095): row = i>>6, c16 = i&63 ; global row stride 1024B, LDS pitch 1040B
  const unsigned h_s_lds = (unsigned)(size_t)(void*)h_s;   // LDS byte offset (low 32 bits)
  __syncthreads();

  for (int t = 0; t < T_STEPS; ++t) {
    // async-stage h (bf16 [64][512]) into padded LDS rows via GLOBAL_LOAD_ASYNC_TO_LDS_B128
#pragma unroll
    for (int q = 0; q < 8; ++q) {
      int i = tid + q * RTHREADS;
      int row = i >> 6, c16 = i & 63;
      async_g2l_b128(h_s_lds + (unsigned)(row * (HPITCH * 2) + c16 * 16),
                     (const char*)h_glob + (size_t)row * (HDIM * 2) + c16 * 16);
    }

    // while the copy flies, prefetch this step's x_proj addends into cache
    const float* xpt = xp + (size_t)t * BATCH * G4H;
    {
      int j0 = r, zc0 = ng * HDIM + wg * 32;
#pragma unroll
      for (int g = 0; g < 8; ++g) {
        int b = mt * 16 + hb * 8 + g;
        __builtin_prefetch(xpt + (size_t)b * G4H + zc0 + j0, 0, 1);
        __builtin_prefetch(xpt + (size_t)b * G4H + zc0 + 16 + j0, 0, 1);
      }
    }

    wait_asynccnt0();
    __syncthreads();

    v8f acc0 = {}, acc1 = {};
#pragma unroll 4
    for (int kk = 0; kk < 16; ++kk) {
      v16bf a  = load_frag_a(h_s, HPITCH, mt * 16, kk * 32, r, hb);
      v16bf b0 = load_frag_b(w_s, WPITCH, ng * 32, kk * 32, r, hb);
      v16bf b1 = load_frag_b(w_s, WPITCH, ng * 32 + 16, kk * 32, r, hb);
      acc0 = wmma_bf16(a, b0, acc0);
      acc1 = wmma_bf16(a, b1, acc1);
    }
    __syncthreads();  // all waves done reading h_s before zs overlay is written

    // add x projection, apply per-gate activation, scatter to zs[gate][b][j]
#pragma unroll
    for (int tj = 0; tj < 2; ++tj) {
      v8f acc = tj ? acc1 : acc0;
      int j = tj * 16 + r;
      int zc = ng * HDIM + wg * 32 + j;
#pragma unroll
      for (int g = 0; g < 8; ++g) {
        int b = mt * 16 + hb * 8 + g;
        float v = acc[g] + xpt[(size_t)b * G4H + zc];
        float act = (ng == 0) ? tanhf(v) : 1.f / (1.f + __expf(-v));
        zs[ng * 2048 + b * 32 + j] = act;
      }
    }
    __syncthreads();

    // combine gates, update c (registers), emit h (f32 out + bf16 h_glob)
#pragma unroll
    for (int q = 0; q < 4; ++q) {
      int e = tid + q * RTHREADS;
      int b = e >> 5, j = e & 31;
      float gg = zs[0 * 2048 + b * 32 + j];
      float ii = zs[1 * 2048 + b * 32 + j];
      float ff = zs[2 * 2048 + b * 32 + j];
      float oo = zs[3 * 2048 + b * 32 + j];
      float c = gg * ii + creg[q] * ff;
      creg[q] = c;
      float h = tanhf(c) * oo;
      out[(size_t)t * BATCH * HDIM + (size_t)b * HDIM + wg * 32 + j] = h;
      h_glob[(size_t)b * HDIM + wg * 32 + j] = (__bf16)h;
    }

    // device-wide barrier across the 16 persistent WGs
    __threadfence();
    __syncthreads();
    if (tid == 0) {
      atomicAdd(bar, 1u);
      const unsigned target = (unsigned)NWG * (unsigned)(t + 1);
      while (__hip_atomic_load(bar, __ATOMIC_ACQUIRE, __HIP_MEMORY_SCOPE_AGENT) < target) {
        __builtin_amdgcn_s_sleep(2);
      }
    }
    __syncthreads();
  }
}

extern "C" void kernel_launch(void* const* d_in, const int* in_sizes, int n_in,
                              void* d_out, int out_size, void* d_ws, size_t ws_size,
                              hipStream_t stream) {
  const float* embeds = (const float*)d_in[0];
  const float* w_gx = (const float*)d_in[1];
  const float* w_gh = (const float*)d_in[2];
  const float* b_g  = (const float*)d_in[3];
  const float* w_ix = (const float*)d_in[4];
  const float* w_ih = (const float*)d_in[5];
  const float* b_i  = (const float*)d_in[6];
  const float* w_fx = (const float*)d_in[7];
  const float* w_fh = (const float*)d_in[8];
  const float* b_f  = (const float*)d_in[9];
  const float* w_ox = (const float*)d_in[10];
  const float* w_oh = (const float*)d_in[11];
  const float* b_o  = (const float*)d_in[12];
  float* out = (float*)d_out;

  // workspace layout
  char* ws = (char*)d_ws;
  unsigned* bar      = (unsigned*)ws;                                  // 256 B slot
  __bf16*   Wx       = (__bf16*)(ws + 256);                            // 2 MB
  __bf16*   Wh       = (__bf16*)(ws + 256 + 2097152);                  // 2 MB
  float*    bias_cat = (float*) (ws + 256 + 2 * 2097152);              // 8 KB
  __bf16*   h_glob   = (__bf16*)(ws + 256 + 2 * 2097152 + 8192);       // 64 KB
  float*    xp       = (float*) (ws + 256 + 2 * 2097152 + 8192 + 65536); // 512 MB

  const size_t prep_total = (size_t)EDIM * G4H + G4H + (size_t)BATCH * HDIM + 1;
  prep_kernel<<<(unsigned)((prep_total + 255) / 256), 256, 0, stream>>>(
      w_gx, w_gh, b_g, w_ix, w_ih, b_i, w_fx, w_fh, b_f, w_ox, w_oh, b_o,
      Wx, Wh, bias_cat, h_glob, bar);

  dim3 gx(G4H / 64, (T_STEPS * BATCH) / 128);
  xproj_kernel<<<gx, 256, 0, stream>>>(embeds, Wx, bias_cat, xp);

  const size_t lds_bytes = (size_t)128 * WPITCH * 2 + (size_t)64 * HPITCH * 2;  // ~195 KB
  (void)hipFuncSetAttribute((const void*)lstm_rec,
                            hipFuncAttributeMaxDynamicSharedMemorySize, (int)lds_bytes);
  lstm_rec<<<NWG, RTHREADS, lds_bytes, stream>>>(Wh, xp, h_glob, out, bar);
}
